// GREATEncoder_83434034692197
// MI455X (gfx1250) — matrix operand order repaired and checked by reading, not compiled, not run on hardware
//
#include <hip/hip_runtime.h>
#include <hip/hip_bf16.h>
#include <math.h>

typedef _Float16 hf;
typedef __attribute__((ext_vector_type(16))) _Float16 v16h;
typedef __attribute__((ext_vector_type(8)))  _Float16 v8h;
typedef __attribute__((ext_vector_type(8)))  float    v8f;

#define EMB   128
#define HEADS 8
#define DH    16
#define FFD   512
#define NB    48
#define NN    100
#define NPAIR 4950      // NN*(NN-1)/2
#define EPP   9900      // directed edges per problem
#define ETOT  475200    // NB*EPP
#define NTOT  4800      // NB*NN
#define BN_EPS 1e-5f

#define USE_ASYNC_LDS 1

// ---------------- WMMA helpers (gfx1250, wave32) ----------------

__device__ __forceinline__ v8f wmma_f16(v16h a, v16h b, v8f c) {
  // v_wmma_f32_16x16x32_f16: D = A(16x32 f16) * B(32x16 f16) + C(16x16 f32)
  return __builtin_amdgcn_wmma_f32_16x16x32_f16(false, a, false, b, (short)0, c, false, false);
}

// A-frag, direct from row-major f16 memory. p = rowPtr + kk + ((lane<16)?0:8):
// elements 0..7 <- p[0..7], 8..15 <- p[16..23]  (ISA 16-bit A 16x32 layout)
__device__ __forceinline__ v16h ld_frag2(const hf* p) {
  v16h f;
#pragma unroll
  for (int i = 0; i < 8; ++i) { f[i] = p[i]; f[8 + i] = p[16 + i]; }
  return f;
}

// B-frag: lane n (0-15) holds col n with K 0..7,16..23; lane n+16 holds K 8..15,24..31.
// p = Wt + n*K + kk + ((lane<16)?0:8): elements 0..7 <- p[0..7], 8..15 <- p[16..23].
__device__ __forceinline__ v16h ld_bfrag(const hf* p) {
  v16h f;
#pragma unroll
  for (int i = 0; i < 8; ++i) { f[i] = p[i]; f[8 + i] = p[16 + i]; }
  return f;
}

// A-frag from LDS with row stride (used by FFN hidden transpose)
__device__ __forceinline__ v16h ld_afrag_lds(const hf* base, int stride, int lane) {
  int r  = lane & 15;
  int ko = (lane < 16) ? 0 : 8;
  const hf* p = base + r * stride + ko;
  v16h f;
#pragma unroll
  for (int i = 0; i < 8; ++i) { f[i] = p[i]; f[8 + i] = p[16 + i]; }
  return f;
}

// ---------------- async global -> LDS copy (CDNA5) ----------------

__device__ __forceinline__ void async_cp16(hf* lds, const hf* g) {
#if USE_ASYNC_LDS
  unsigned lo = (unsigned)(size_t)lds; // low 32 bits of the LDS pointer, HW adds wave LDS base
  asm volatile("global_load_async_to_lds_b128 %0, %1, off"
               :: "v"(lo), "v"(g) : "memory");
#else
  *(v8h*)lds = *(const v8h*)g;
#endif
}

__device__ __forceinline__ void wait_async0() {
#if USE_ASYNC_LDS
  asm volatile("s_wait_asynccnt 0x0" ::: "memory");
#endif
}

// ---------------- setup kernels ----------------

__global__ void k_build_pairs(int* s_idx, int* d_idx) {
  int t = blockIdx.x * blockDim.x + threadIdx.x;
  if (t >= NPAIR) return;
  int i = 0, rem = t;
  while (rem >= NN - 1 - i) { rem -= NN - 1 - i; ++i; }
  int j = i + 1 + rem;
  s_idx[2 * t]     = i; d_idx[2 * t]     = j;
  s_idx[2 * t + 1] = j; d_idx[2 * t + 1] = i;
}

__global__ void k_build_inc(const int* __restrict__ d_idx, int* __restrict__ inc) {
  int j = threadIdx.x;
  if (j >= NN) return;
  int c = 0;
  for (int e = 0; e < EPP; ++e)
    if (d_idx[e] == j) inc[j * 99 + (c++)] = e;
}

// W (din x dout, row-major f32) -> Wt (dout x din, f16)
__global__ void k_conv_wt(const float* __restrict__ W, hf* __restrict__ Wt, int din, int dout) {
  int i = blockIdx.x * blockDim.x + threadIdx.x;
  if (i >= din * dout) return;
  int k = i / dout, n = i - k * dout;
  Wt[(size_t)n * din + k] = (hf)W[i];
}

__global__ void k_embed(const float* __restrict__ problems, const float* __restrict__ eW,
                        const float* __restrict__ eb, const int* __restrict__ s_idx,
                        const int* __restrict__ d_idx, hf* __restrict__ edges) {
  long idx = (long)blockIdx.x * blockDim.x + threadIdx.x;
  if (idx >= (long)ETOT * 128) return;
  int e = (int)(idx >> 7), c = (int)(idx & 127);
  int b = e / EPP, t2 = e - b * EPP;
  int s = s_idx[t2], d = d_idx[t2];
  const float* pp = problems + (((size_t)b * NN + s) * NN + d) * 2;
  edges[idx] = (hf)(pp[0] * eW[c] + pp[1] * eW[128 + c] + eb[c]);
}

// ---------------- GEMM: C[M x 128] = A[M x K] @ W[K x 128] + bias (+resid), all-f16 I/O ----------------
// No LDS, no barriers: A-frags stream straight from row-major f16 global memory.

__global__ void k_gemm128(const hf* __restrict__ A, const hf* __restrict__ Wt,
                          const float* __restrict__ bias, const hf* __restrict__ resid,
                          hf* __restrict__ C, int M, int K) {
  int tid = threadIdx.x, lane = tid & 31, w = tid >> 5;
  int row0 = blockIdx.x * 128, wr = w * 16;
  int nb = lane & 15, kob = (lane < 16) ? 0 : 8, alo = (lane < 16) ? 0 : 8;
  int arow = row0 + wr + (lane & 15);
  if (arow >= M) arow = M - 1;                 // clamp: stays in-buffer, store is guarded
  const hf* arp = A + (size_t)arow * K;
  v8f acc[8] = {};
  for (int kk = 0; kk < K; kk += 32) {
    v16h af = ld_frag2(arp + kk + alo);
#pragma unroll
    for (int nt = 0; nt < 8; ++nt) {
      v16h bf = ld_bfrag(Wt + (size_t)(nt * 16 + nb) * K + kk + kob);
      acc[nt] = wmma_f16(af, bf, acc[nt]);
    }
  }
#pragma unroll
  for (int nt = 0; nt < 8; ++nt) {
    int n = nt * 16 + nb;
#pragma unroll
    for (int vv = 0; vv < 8; ++vv) {
      int m = row0 + wr + vv + ((lane >> 4) << 3);
      if (m < M) {
        float o = acc[nt][vv] + bias[n];
        if (resid) o += (float)resid[(size_t)m * 128 + n];
        C[(size_t)m * 128 + n] = (hf)o;
      }
    }
  }
}

// ---------------- fused QK projection + per-head dot (alpha) ----------------

__global__ void k_qk_alpha(const hf* __restrict__ A, const hf* __restrict__ Wqt,
                           const float* __restrict__ bq, const hf* __restrict__ Wkt,
                           const float* __restrict__ bk, float* __restrict__ alpha, int M) {
  int tid = threadIdx.x, lane = tid & 31, w = tid >> 5;
  int row0 = blockIdx.x * 128, wr = w * 16;
  int nb = lane & 15, kob = (lane < 16) ? 0 : 8, alo = (lane < 16) ? 0 : 8;
  int arow = row0 + wr + (lane & 15);
  if (arow >= M) arow = M - 1;
  const hf* arp = A + (size_t)arow * 128;
  v8f qa[8] = {}, ka[8] = {};
  for (int kk = 0; kk < 128; kk += 32) {
    v16h af = ld_frag2(arp + kk + alo);
#pragma unroll
    for (int nt = 0; nt < 8; ++nt) {
      v16h bfq = ld_bfrag(Wqt + (size_t)(nt * 16 + nb) * 128 + kk + kob);
      qa[nt] = wmma_f16(af, bfq, qa[nt]);
      v16h bfk = ld_bfrag(Wkt + (size_t)(nt * 16 + nb) * 128 + kk + kob);
      ka[nt] = wmma_f16(af, bfk, ka[nt]);
    }
  }
  // head h == column tile nt (DH==16). Reduce q*k across the 16 lanes of each half.
#pragma unroll
  for (int nt = 0; nt < 8; ++nt) {
    int n = nt * 16 + nb;
    float bqv = bq[n], bkv = bk[n];
#pragma unroll
    for (int vv = 0; vv < 8; ++vv) {
      float p = (qa[nt][vv] + bqv) * (ka[nt][vv] + bkv);
      p += __shfl_xor(p, 1);
      p += __shfl_xor(p, 2);
      p += __shfl_xor(p, 4);
      p += __shfl_xor(p, 8);
      int m = row0 + wr + vv + ((lane >> 4) << 3);
      if (((lane & 15) == 0) && m < M) alpha[(size_t)m * 8 + nt] = p * 0.25f; // 1/sqrt(16)
    }
  }
}

// ---------------- segment softmax + aggregation; v rows gathered via async->LDS ----------------

__global__ void k_attn(const float* __restrict__ alpha, const hf* __restrict__ v,
                       const int* __restrict__ inc, hf* __restrict__ agg) {
  __shared__ float a_s[99 * 8];
  __shared__ float w_s[99 * 8];
  __shared__ int   e_s[99];
  __shared__ __align__(16) hf v_s[99 * 128];
  int nd = blockIdx.x, tid = threadIdx.x;
  int b = nd / NN, j = nd - b * NN;
  if (tid < 99) {
    int el = inc[j * 99 + tid];
    e_s[tid] = el;
    int eg = b * EPP + el;
#pragma unroll
    for (int h = 0; h < 8; ++h) a_s[tid * 8 + h] = alpha[(size_t)eg * 8 + h];
  }
  __syncthreads();
  // issue async copies of the 99 scattered v rows (99*16 chunks of 16B)
  for (int li = tid; li < 99 * 16; li += 128) {
    int i = li >> 4, c = li & 15;
    async_cp16(&v_s[i * 128 + c * 8], v + (size_t)(b * EPP + e_s[i]) * 128 + c * 8);
  }
  // softmax overlaps the async DMA
  if (tid < 8) {
    float m = -3.4e38f;
    for (int i = 0; i < 99; ++i) m = fmaxf(m, a_s[i * 8 + tid]);
    float s = 0.f;
    for (int i = 0; i < 99; ++i) s += expf(a_s[i * 8 + tid] - m);
    float inv = 1.f / (s + 1e-16f);
    for (int i = 0; i < 99; ++i) w_s[i * 8 + tid] = expf(a_s[i * 8 + tid] - m) * inv;
  }
  wait_async0();
  __syncthreads();
  int h = tid >> 4;
  float acc = 0.f;
  for (int i = 0; i < 99; ++i)
    acc += w_s[i * 8 + h] * (float)v_s[i * 128 + tid];
  agg[(size_t)nd * 128 + tid] = (hf)acc;
}

// ---------------- gather-GEMM: tmp = [out[src] || out[dst]] @ We + be + edges ----------------

__global__ void k_eagg(const hf* __restrict__ outn, const int* __restrict__ s_idx,
                       const int* __restrict__ d_idx, const hf* __restrict__ Wet,
                       const float* __restrict__ be, const hf* __restrict__ resid,
                       hf* __restrict__ C) {
  int tid = threadIdx.x, lane = tid & 31, w = tid >> 5;
  int row0 = blockIdx.x * 128, wr = w * 16;
  int nb = lane & 15, kob = (lane < 16) ? 0 : 8, alo = (lane < 16) ? 0 : 8;
  int e = row0 + wr + (lane & 15);
  if (e >= ETOT) e = ETOT - 1;
  int b = e / EPP, t2 = e - b * EPP;
  const hf* rs = outn + (size_t)(b * NN + s_idx[t2]) * 128; // first 128 k: out[src]
  const hf* rd = outn + (size_t)(b * NN + d_idx[t2]) * 128; // next 128 k:  out[dst]
  v8f acc[8] = {};
  for (int kk = 0; kk < 256; kk += 32) {
    const hf* ap = (kk < 128) ? (rs + kk) : (rd + kk - 128);
    v16h af = ld_frag2(ap + alo);
#pragma unroll
    for (int nt = 0; nt < 8; ++nt) {
      v16h bf = ld_bfrag(Wet + (size_t)(nt * 16 + nb) * 256 + kk + kob);
      acc[nt] = wmma_f16(af, bf, acc[nt]);
    }
  }
#pragma unroll
  for (int nt = 0; nt < 8; ++nt) {
    int n = nt * 16 + nb;
#pragma unroll
    for (int vv = 0; vv < 8; ++vv) {
      int m = row0 + wr + vv + ((lane >> 4) << 3);
      if (m < ETOT)
        C[(size_t)m * 128 + n] = (hf)(acc[nt][vv] + be[n] + (float)resid[(size_t)m * 128 + n]);
    }
  }
}

// ---------------- fused FFN: Y = relu(X@W1+b1)@W2 + b2 + resid (hidden stays in LDS) ----------------

__global__ void k_ffn(const hf* __restrict__ X, const hf* __restrict__ W1t,
                      const float* __restrict__ b1, const hf* __restrict__ W2t,
                      const float* __restrict__ b2, const hf* __restrict__ resid,
                      hf* __restrict__ Y, int M) {
  __shared__ hf sh[128 * 136]; // hidden chunk, padded stride
  int tid = threadIdx.x, lane = tid & 31, w = tid >> 5;
  int row0 = blockIdx.x * 128, wr = w * 16;
  int nb = lane & 15, kob = (lane < 16) ? 0 : 8, alo = (lane < 16) ? 0 : 8;
  int arow = row0 + wr + (lane & 15);
  if (arow >= M) arow = M - 1;
  const hf* arp = X + (size_t)arow * 128;
  v16h ax[4];
#pragma unroll
  for (int kk = 0; kk < 4; ++kk) ax[kk] = ld_frag2(arp + kk * 32 + alo);
  v8f oacc[8] = {};
  for (int ch = 0; ch < 4; ++ch) {
    v8f hacc[8] = {};
#pragma unroll
    for (int nt = 0; nt < 8; ++nt)
#pragma unroll
      for (int kk = 0; kk < 4; ++kk) {
        v16h bf = ld_bfrag(W1t + (size_t)(ch * 128 + nt * 16 + nb) * 128 + kk * 32 + kob);
        hacc[nt] = wmma_f16(ax[kk], bf, hacc[nt]);
      }
    // relu + bias, park hidden chunk in LDS (D-layout -> row-major, wave-private rows)
#pragma unroll
    for (int nt = 0; nt < 8; ++nt) {
      int n = nt * 16 + nb;
#pragma unroll
      for (int vv = 0; vv < 8; ++vv) {
        int m = wr + vv + ((lane >> 4) << 3);
        sh[m * 136 + n] = (hf)fmaxf(hacc[nt][vv] + b1[ch * 128 + n], 0.f);
      }
    }
    __syncthreads();
    v16h ah[4];
#pragma unroll
    for (int kk = 0; kk < 4; ++kk) ah[kk] = ld_afrag_lds(&sh[wr * 136 + kk * 32], 136, lane);
#pragma unroll
    for (int nt = 0; nt < 8; ++nt)
#pragma unroll
      for (int kk = 0; kk < 4; ++kk) {
        v16h bf = ld_bfrag(W2t + (size_t)(nt * 16 + nb) * 512 + ch * 128 + kk * 32 + kob);
        oacc[nt] = wmma_f16(ah[kk], bf, oacc[nt]);
      }
    __syncthreads();
  }
#pragma unroll
  for (int nt = 0; nt < 8; ++nt) {
    int n = nt * 16 + nb;
#pragma unroll
    for (int vv = 0; vv < 8; ++vv) {
      int m = row0 + wr + vv + ((lane >> 4) << 3);
      if (m < M)
        Y[(size_t)m * 128 + n] = (hf)(oacc[nt][vv] + b2[n] + (float)resid[(size_t)m * 128 + n]);
    }
  }
}

// ---------------- BatchNorm (f32 column stats over M rows of f16 data) ----------------

__global__ void k_bn_zero(float* s, float* q) {
  int t = threadIdx.x;
  s[t] = 0.f; q[t] = 0.f;
}

__global__ void k_bn_stats(const hf* __restrict__ x, int M, float* s, float* q) {
  int c = threadIdx.x;
  int r0 = blockIdx.x * 512;
  int r1 = r0 + 512; if (r1 > M) r1 = M;
  float ls = 0.f, lq = 0.f;
  for (int r = r0; r < r1; ++r) { float v = (float)x[(size_t)r * 128 + c]; ls += v; lq += v * v; }
  atomicAdd(&s[c], ls);
  atomicAdd(&q[c], lq);
}

__global__ void k_bn_apply_h(const hf* __restrict__ x, hf* __restrict__ y,
                             const float* __restrict__ g, const float* __restrict__ b,
                             const float* __restrict__ s, const float* __restrict__ q,
                             int M, long total) {
  long i = (long)blockIdx.x * blockDim.x + threadIdx.x;
  if (i >= total) return;
  int c = (int)(i & 127);
  float mean = s[c] / (float)M;
  float var  = q[c] / (float)M - mean * mean;
  y[i] = (hf)(((float)x[i] - mean) * rsqrtf(var + BN_EPS) * g[c] + b[c]);
}

__global__ void k_bn_apply_f(const hf* __restrict__ x, float* __restrict__ y,
                             const float* __restrict__ g, const float* __restrict__ b,
                             const float* __restrict__ s, const float* __restrict__ q,
                             int M, long total) {
  long i = (long)blockIdx.x * blockDim.x + threadIdx.x;
  if (i >= total) return;
  int c = (int)(i & 127);
  float mean = s[c] / (float)M;
  float var  = q[c] / (float)M - mean * mean;
  y[i] = ((float)x[i] - mean) * rsqrtf(var + BN_EPS) * g[c] + b[c];
}

// ---------------- host orchestration ----------------

struct Lay {
  const float *Wq,*bq,*Wk,*bk,*Wv,*bv,*Wo,*bo,*We,*be,*W1,*b1,*W2,*b2,*ga,*ba,*gf,*bf;
};

extern "C" void kernel_launch(void* const* d_in, const int* in_sizes, int n_in,
                              void* d_out, int out_size, void* d_ws, size_t ws_size,
                              hipStream_t stream) {
  (void)out_size; (void)ws_size; (void)n_in;
  const float* problems; const float *eW, *eb;
  Lay L[3];
  auto F = [&](int i) { return (const float*)d_in[i]; };
  if (in_sizes[0] == NB * NN * NN * 2) {
    // insertion (dict) order: problems, embed{W,b}, layers[...18 leaves each]
    problems = F(0); eW = F(1); eb = F(2);
    for (int l = 0; l < 3; ++l) {
      int b = 3 + l * 18;
      L[l] = { F(b+0),F(b+1),F(b+2),F(b+3),F(b+4),F(b+5),F(b+6),F(b+7),
               F(b+8),F(b+9),F(b+10),F(b+11),F(b+12),F(b+13),
               F(b+14),F(b+15),F(b+16),F(b+17) };
    }
  } else {
    // JAX sorted-pytree order: params first ('params' < 'problems'), dict keys sorted
    eW = F(0); eb = F(1);
    for (int l = 0; l < 3; ++l) {
      int b = 2 + l * 18;
      // W1,W2,We,Wk,Wo,Wq,Wv,b1,b2,be,bk,bn_att_b,bn_att_g,bn_ff_b,bn_ff_g,bo,bq,bv
      L[l].W1 = F(b+0);  L[l].W2 = F(b+1);  L[l].We = F(b+2);
      L[l].Wk = F(b+3);  L[l].Wo = F(b+4);  L[l].Wq = F(b+5);  L[l].Wv = F(b+6);
      L[l].b1 = F(b+7);  L[l].b2 = F(b+8);  L[l].be = F(b+9);  L[l].bk = F(b+10);
      L[l].ba = F(b+11); L[l].ga = F(b+12); L[l].bf = F(b+13); L[l].gf = F(b+14);
      L[l].bo = F(b+15); L[l].bq = F(b+16); L[l].bv = F(b+17);
    }
    problems = F(2 + 3 * 18);
  }

  char* ws = (char*)d_ws;
  size_t off = 0;
  auto take = [&](size_t bytes) -> void* {
    void* p = ws + off;
    off = (off + bytes + 255) & ~(size_t)255;
    return p;
  };
  int* sIdx = (int*)take(EPP * 4);
  int* dIdx = (int*)take(EPP * 4);
  int* inc  = (int*)take(NN * 99 * 4);
  hf *wq16[3], *wk16[3], *wv16[3], *wo16[3], *we16[3], *w116[3], *w216[3];
  for (int l = 0; l < 3; ++l) {
    wq16[l] = (hf*)take(128 * 128 * 2);
    wk16[l] = (hf*)take(128 * 128 * 2);
    wv16[l] = (hf*)take(128 * 128 * 2);
    wo16[l] = (hf*)take(128 * 128 * 2);
    we16[l] = (hf*)take(256 * 128 * 2);
    w116[l] = (hf*)take((size_t)128 * 512 * 2);
    w216[l] = (hf*)take((size_t)512 * 128 * 2);
  }
  float* alpha = (float*)take((size_t)ETOT * 8 * 4);
  hf* agg   = (hf*)take((size_t)NTOT * 128 * 2);
  hf* bout  = (hf*)take((size_t)NTOT * 128 * 2);
  hf* ntmp  = (hf*)take((size_t)NTOT * 128 * 2);
  float* bsum = (float*)take(128 * 4);
  float* bsq  = (float*)take(128 * 4);
  hf* edges = (hf*)take((size_t)ETOT * 128 * 2);
  hf* tmp   = (hf*)take((size_t)ETOT * 128 * 2);

  k_build_pairs<<<(NPAIR + 255) / 256, 256, 0, stream>>>(sIdx, dIdx);
  k_build_inc<<<1, 128, 0, stream>>>(dIdx, inc);
  for (int l = 0; l < 3; ++l) {
    k_conv_wt<<<64, 256, 0, stream>>>(L[l].Wq, wq16[l], 128, 128);
    k_conv_wt<<<64, 256, 0, stream>>>(L[l].Wk, wk16[l], 128, 128);
    k_conv_wt<<<64, 256, 0, stream>>>(L[l].Wv, wv16[l], 128, 128);
    k_conv_wt<<<64, 256, 0, stream>>>(L[l].Wo, wo16[l], 128, 128);
    k_conv_wt<<<128, 256, 0, stream>>>(L[l].We, we16[l], 256, 128);
    k_conv_wt<<<256, 256, 0, stream>>>(L[l].W1, w116[l], 128, 512);
    k_conv_wt<<<256, 256, 0, stream>>>(L[l].W2, w216[l], 512, 128);
  }
  {
    long tot = (long)ETOT * 128;
    k_embed<<<(int)((tot + 255) / 256), 256, 0, stream>>>(problems, eW, eb, sIdx, dIdx, edges);
  }

  const int GBE = (ETOT + 127) / 128;
  const int GBN = (NTOT + 127) / 128;

  auto bn_h = [&](const hf* x, hf* y, const float* g, const float* b, int M) {
    k_bn_zero<<<1, 128, 0, stream>>>(bsum, bsq);
    k_bn_stats<<<(M + 511) / 512, 128, 0, stream>>>(x, M, bsum, bsq);
    long tot = (long)M * 128;
    k_bn_apply_h<<<(int)((tot + 255) / 256), 256, 0, stream>>>(x, y, g, b, bsum, bsq, M, tot);
  };

  for (int l = 0; l < 3; ++l) {
    const Lay& p = L[l];
    k_gemm128<<<GBE, 256, 0, stream>>>(edges, wv16[l], p.bv, nullptr, tmp, ETOT, 128);   // V
    k_qk_alpha<<<GBE, 256, 0, stream>>>(edges, wq16[l], p.bq, wk16[l], p.bk, alpha, ETOT);
    k_attn<<<NTOT, 128, 0, stream>>>(alpha, tmp, inc, agg);
    k_gemm128<<<GBN, 256, 0, stream>>>(agg, wo16[l], p.bo, nullptr, bout, NTOT, 128);    // out
    if (l < 2) {
      k_eagg<<<GBE, 256, 0, stream>>>(bout, sIdx, dIdx, we16[l], p.be, edges, tmp);
      bn_h(tmp, edges, p.ga, p.ba, ETOT);
      k_ffn<<<GBE, 256, 0, stream>>>(edges, w116[l], p.b1, w216[l], p.b2, edges, tmp, ETOT);
      bn_h(tmp, edges, p.gf, p.bf, ETOT);
    } else {
      bn_h(bout, ntmp, p.ga, p.ba, NTOT);
      k_ffn<<<GBN, 256, 0, stream>>>(ntmp, w116[l], p.b1, w216[l], p.b2, ntmp, agg, NTOT);
      k_bn_zero<<<1, 128, 0, stream>>>(bsum, bsq);
      k_bn_stats<<<(NTOT + 511) / 512, 128, 0, stream>>>(agg, NTOT, bsum, bsq);
      long tot = (long)NTOT * 128;
      k_bn_apply_f<<<(int)((tot + 255) / 256), 256, 0, stream>>>(agg, (float*)d_out,
                                                                 p.gf, p.bf, bsum, bsq, NTOT, tot);
    }
  }
}
